// GNNImageClassificator_21680994910456
// MI455X (gfx1250) — matrix operand order repaired and checked by reading, not compiled, not run on hardware
//
#include <hip/hip_runtime.h>
#include <hip/hip_bf16.h>

typedef __attribute__((ext_vector_type(16))) __bf16 v16bf;
typedef __attribute__((ext_vector_type(8)))  __bf16 v8bf;
typedef __attribute__((ext_vector_type(8)))  float  v8f;
typedef __attribute__((ext_vector_type(4)))  unsigned int v4u;
typedef __attribute__((ext_vector_type(8)))  int    v8i_t;
typedef __attribute__((ext_vector_type(4)))  int    v4i_t;

#define BB   64
#define NN   1024
#define EE   16384
#define HH   152
#define HPAD 160
#define EN   (EE + NN)      // 17408
#define OFFS (NN + 1)       // 1025

#if __has_include(<hip/amd_detail/amd_gfx1250_TDM.h>)
#define TDM_6ARG 1
#endif

__device__ __forceinline__ v8bf ld8(const __bf16* p) {
  return *reinterpret_cast<const v8bf*>(p);
}

// ---------------------------------------------------------------- CSR build
__global__ void k_init_counts(int* counts) {
  int i = blockIdx.x * blockDim.x + threadIdx.x;           // B*N
  if (i < BB * NN) counts[i] = 1;                          // self-loop
}

__global__ void k_count_edges(const int* ei, int* counts) {
  int i = blockIdx.x * blockDim.x + threadIdx.x;           // B*E
  if (i >= BB * EE) return;
  int b = i / EE, e = i % EE;
  int d = ei[(size_t)b * 2 * EE + EE + e];
  atomicAdd(&counts[b * NN + d], 1);
}

__global__ void k_scan(const int* counts, int* offsets, int* cursor) {
  __shared__ int s[NN];
  int b = blockIdx.x, t = threadIdx.x;                     // 1024 threads
  s[t] = counts[b * NN + t];
  __syncthreads();
  for (int off = 1; off < NN; off <<= 1) {
    int v = (t >= off) ? s[t - off] : 0;
    __syncthreads();
    s[t] += v;
    __syncthreads();
  }
  int excl = (t == 0) ? 0 : s[t - 1];
  offsets[b * OFFS + t] = excl;
  cursor[b * NN + t] = excl;
  if (t == NN - 1) offsets[b * OFFS + NN] = s[NN - 1];
}

__global__ void k_fill_self(int* cursor, int* srcl) {
  int i = blockIdx.x * blockDim.x + threadIdx.x;           // B*N
  if (i >= BB * NN) return;
  int b = i / NN, n = i % NN;
  int pos = atomicAdd(&cursor[b * NN + n], 1);
  srcl[(size_t)b * EN + pos] = n;
}

__global__ void k_fill_edges(const int* ei, int* cursor, int* srcl) {
  int i = blockIdx.x * blockDim.x + threadIdx.x;           // B*E
  if (i >= BB * EE) return;
  int b = i / EE, e = i % EE;
  int sN = ei[(size_t)b * 2 * EE + e];
  int d  = ei[(size_t)b * 2 * EE + EE + e];
  int pos = atomicAdd(&cursor[b * NN + d], 1);
  srcl[(size_t)b * EN + pos] = sN;
}

// ------------------------------------------------------- weight / A staging
// Wt[n][k] = W[k][n], zero-padded to [HPAD][Kpad], bf16
__global__ void k_conv_w(const float* W, __bf16* Wt, int K, int Kpad) {
  int i = blockIdx.x * blockDim.x + threadIdx.x;
  if (i >= HPAD * Kpad) return;
  int k = i % Kpad, n = i / Kpad;
  float v = (k < K && n < HH) ? W[k * HH + n] : 0.0f;
  Wt[i] = (__bf16)v;
}

// mode 0: x[.,3] -> [.,32] ; mode 1: x1[.,152] -> [.,160] ; mode 2: concat(x,x2)[.,155] -> [.,160]
__global__ void k_stage(const float* x, const float* xa, __bf16* Abuf, int mode, int Kpad) {
  size_t i = (size_t)blockIdx.x * blockDim.x + threadIdx.x;
  size_t total = (size_t)BB * NN * Kpad;
  if (i >= total) return;
  int k = (int)(i % Kpad);
  size_t node = i / Kpad;                                  // b*N + n
  float v = 0.0f;
  if (mode == 0) {
    if (k < 3) v = x[node * 3 + k];
  } else if (mode == 1) {
    if (k < HH) v = xa[node * HH + k];
  } else {
    if (k < 3) v = x[node * 3 + k];
    else if (k < 155) v = xa[node * HH + (k - 3)];
  }
  Abuf[i] = (__bf16)v;
}

// ------------------------------------------------------------ WMMA GEMM
// Each block (8 waves): weights TDM'd into LDS once; each wave computes a
// 16 x 160 output strip (one A fragment feeds 10 WMMAs per K-tile).
__global__ void __launch_bounds__(256) k_gemm(const __bf16* __restrict__ A,
                                              const __bf16* __restrict__ Wt,
                                              float* __restrict__ Hout,
                                              int Kpad, int Ktiles) {
  __shared__ __bf16 sWt[HPAD * HPAD];                      // 51200 B max
  const int lane = threadIdx.x & 31;
  const int wid  = threadIdx.x >> 5;
  const int b    = blockIdx.x >> 3;
  const int mt   = ((blockIdx.x & 7) << 3) | wid;          // 0..63

  // ---- stage weights into LDS
#if __has_builtin(__builtin_amdgcn_tensor_load_to_lds)
  if (threadIdx.x < 32) {                                  // wave 0 issues TDM
    unsigned ndw = 80u * (unsigned)Kpad;                   // HPAD*Kpad*2/4 dwords
    unsigned long long ga = (unsigned long long)(uintptr_t)Wt;
    unsigned ldsa = (unsigned)(uintptr_t)&sWt[0];
    v4u g0;
    g0[0] = 1u;                                            // count=1, user mode
    g0[1] = ldsa;                                          // lds_addr
    g0[2] = (unsigned)(ga & 0xFFFFFFFFu);                  // global_addr lo
    g0[3] = (unsigned)((ga >> 32) & 0x1FFFFFFu) | 0x80000000u; // addr hi | type=2
    v8i_t g1;
    g1[0] = (int)(2u << 16);                               // data_size = 4B
    g1[1] = (int)((ndw & 0xFFFFu) << 16);                  // tensor_dim0 lo16
    g1[2] = (int)(((ndw >> 16) & 0xFFFFu) | (1u << 16));   // dim0 hi16 | tensor_dim1=1
    g1[3] = (int)(ndw << 16);                              // tile_dim0 = ndw
    g1[4] = 0;                                             // tile_dim1/2 unused
    g1[5] = (int)ndw;                                      // tensor_dim0_stride lo32
    g1[6] = 0;
    g1[7] = 0;
    v4i_t g2 = {0, 0, 0, 0}, g3 = {0, 0, 0, 0};
#ifdef TDM_6ARG
    v8i_t g4 = {0, 0, 0, 0, 0, 0, 0, 0};
    __builtin_amdgcn_tensor_load_to_lds(g0, g1, g2, g3, g4, 0);
#else
    __builtin_amdgcn_tensor_load_to_lds(g0, g1, g2, g3, 0);
#endif
    __builtin_amdgcn_s_wait_tensorcnt(0);
  }
#else
  {
    const unsigned* src = (const unsigned*)Wt;
    unsigned* dst = (unsigned*)&sWt[0];
    int ndw = 80 * Kpad;
    for (int i = threadIdx.x; i < ndw; i += 256) dst[i] = src[i];
  }
#endif
  __syncthreads();

  const __bf16* Ab = A + ((size_t)b * NN + (size_t)mt * 16) * Kpad;
  const int m    = lane & 15;
  const int half = lane >> 4;
  const int aoff = half ? 16 : 0;
  const int ncol = lane & 15;
  const int boff = half ? 16 : 0;

  v8f acc[10];
#pragma unroll
  for (int nt = 0; nt < 10; ++nt)
    acc[nt] = (v8f){0.f, 0.f, 0.f, 0.f, 0.f, 0.f, 0.f, 0.f};

  for (int kt = 0; kt < Ktiles; ++kt) {
    int kb = kt * 32;
    v8bf alo = ld8(Ab + (size_t)m * Kpad + kb + aoff);
    v8bf ahi = ld8(Ab + (size_t)m * Kpad + kb + aoff + 8);
    v16bf av = __builtin_shufflevector(alo, ahi, 0,1,2,3,4,5,6,7,8,9,10,11,12,13,14,15);
#pragma unroll
    for (int nt = 0; nt < 10; ++nt) {
      const __bf16* bp = &sWt[(size_t)(nt * 16 + ncol) * Kpad + kb + boff];
      v8bf blo = ld8(bp);
      v8bf bhi = ld8(bp + 8);
      v16bf bv = __builtin_shufflevector(blo, bhi, 0,1,2,3,4,5,6,7,8,9,10,11,12,13,14,15);
      acc[nt] = __builtin_amdgcn_wmma_f32_16x16x32_bf16(false, av, false, bv,
                                                        (short)0, acc[nt], false, false);
    }
  }

  const int rbase = mt * 16 + half * 8;
#pragma unroll
  for (int nt = 0; nt < 10; ++nt) {
    int col = nt * 16 + ncol;
    if (col < HH) {
#pragma unroll
      for (int r = 0; r < 8; ++r)
        Hout[((size_t)b * NN + rbase + r) * HH + col] = acc[nt][r];
    }
  }
}

// ------------------------------------------------------- attention scores
__global__ void k_scores(const float* __restrict__ Hm, const float* asrc, const float* adst,
                         float* hs, float* hd) {
  int wave = blockIdx.x * (blockDim.x >> 5) + (threadIdx.x >> 5);   // b*N+n
  int lane = threadIdx.x & 31;
  if (wave >= BB * NN) return;
  const float* row = Hm + (size_t)wave * HH;
  float ss = 0.f, sd = 0.f;
  for (int k = lane; k < HH; k += 32) {
    float v = row[k];
    ss += v * asrc[k];
    sd += v * adst[k];
  }
  for (int o = 16; o > 0; o >>= 1) {
    ss += __shfl_xor(ss, o, 32);
    sd += __shfl_xor(sd, o, 32);
  }
  if (lane == 0) { hs[wave] = ss; hd[wave] = sd; }
}

// ----------------------------------------------- softmax-weighted aggregate
__global__ void k_aggregate(const float* __restrict__ Hm, const float* __restrict__ hs,
                            const float* __restrict__ hd, const int* __restrict__ offsets,
                            const int* __restrict__ srcl, const float* bias,
                            float* __restrict__ out) {
  int wave = blockIdx.x * (blockDim.x >> 5) + (threadIdx.x >> 5);
  int lane = threadIdx.x & 31;
  if (wave >= BB * NN) return;
  int b = wave >> 10, d = wave & (NN - 1);
  int base = b * NN;
  int o0 = offsets[b * OFFS + d], o1 = offsets[b * OFFS + d + 1];
  const int* sl = srcl + (size_t)b * EN;
  float hdv = hd[base + d];

  float mx = -3.0e38f;
  for (int j = o0 + lane; j < o1; j += 32) {
    float e = hs[base + sl[j]] + hdv;
    e = (e < 0.f) ? 0.2f * e : e;
    mx = fmaxf(mx, e);
  }
  for (int o = 16; o > 0; o >>= 1) mx = fmaxf(mx, __shfl_xor(mx, o, 32));

  float den = 0.f;
  for (int j = o0 + lane; j < o1; j += 32) {
    float e = hs[base + sl[j]] + hdv;
    e = (e < 0.f) ? 0.2f * e : e;
    den += __expf(e - mx);
  }
  for (int o = 16; o > 0; o >>= 1) den += __shfl_xor(den, o, 32);
  float inv = 1.0f / fmaxf(den, 1e-16f);

  float acc[5] = {0.f, 0.f, 0.f, 0.f, 0.f};
  for (int j = o0; j < o1; ++j) {
    int sN = sl[j];
    if (j + 1 < o1)                                        // prefetch next row
      __builtin_prefetch(Hm + (size_t)(base + sl[j + 1]) * HH + lane, 0, 1);
    float e = hs[base + sN] + hdv;
    e = (e < 0.f) ? 0.2f * e : e;
    float w = __expf(e - mx);
    const float* row = Hm + (size_t)(base + sN) * HH;
#pragma unroll
    for (int c = 0; c < 5; ++c) {
      int k = lane + 32 * c;
      if (k < HH) acc[c] += w * row[k];
    }
  }
#pragma unroll
  for (int c = 0; c < 5; ++c) {
    int k = lane + 32 * c;
    if (k < HH) out[(size_t)(base + d) * HH + k] = acc[c] * inv + bias[k];
  }
}

// --------------------------------------------------- mean-pool + MLP head
__global__ void k_head(const float* x, const float* x1, const float* x2, const float* x3,
                       const float* fw1, const float* fb1, const float* fw2, const float* fb2,
                       const float* fw3, const float* fb3, const float* fw4, const float* fb4,
                       float* out) {
  __shared__ float feat[459];
  __shared__ float l1[256];
  __shared__ float l2[32];
  __shared__ float l3[32];
  int b = blockIdx.x, t = threadIdx.x;

  for (int f = t; f < 459; f += 256) {
    float s = 0.f;
    if (f < 3) {
      const float* p = x + (size_t)b * NN * 3 + f;
      for (int n = 0; n < NN; ++n) s += p[(size_t)n * 3];
    } else if (f < 155) {
      const float* p = x1 + (size_t)b * NN * HH + (f - 3);
      for (int n = 0; n < NN; ++n) s += p[(size_t)n * HH];
    } else if (f < 307) {
      const float* p = x2 + (size_t)b * NN * HH + (f - 155);
      for (int n = 0; n < NN; ++n) s += p[(size_t)n * HH];
    } else {
      const float* p = x3 + (size_t)b * NN * HH + (f - 307);
      for (int n = 0; n < NN; ++n) s += p[(size_t)n * HH];
    }
    feat[f] = s * (1.0f / NN);
  }
  __syncthreads();
  {
    float s = fb1[t];
    for (int k = 0; k < 459; ++k) s += feat[k] * fw1[k * 256 + t];
    l1[t] = fmaxf(s, 0.f);
  }
  __syncthreads();
  if (t < 32) {
    float s = fb2[t];
    for (int k = 0; k < 256; ++k) s += l1[k] * fw2[k * 32 + t];
    l2[t] = fmaxf(s, 0.f);
  }
  __syncthreads();
  if (t < 32) {
    float s = fb3[t];
    for (int k = 0; k < 32; ++k) s += l2[k] * fw3[k * 32 + t];
    l3[t] = fmaxf(s, 0.f);
  }
  __syncthreads();
  if (t < 10) {
    float s = fb4[t];
    for (int k = 0; k < 32; ++k) s += l3[k] * fw4[k * 10 + t];
    out[b * 10 + t] = s;
  }
}

// ---------------------------------------------------------------- launcher
extern "C" void kernel_launch(void* const* d_in, const int* in_sizes, int n_in,
                              void* d_out, int out_size, void* d_ws, size_t ws_size,
                              hipStream_t stream) {
  (void)in_sizes; (void)n_in; (void)out_size; (void)ws_size;
  const float* x   = (const float*)d_in[0];
  const int*   ei  = (const int*)d_in[1];
  const float* W[3]    = {(const float*)d_in[2],  (const float*)d_in[6],  (const float*)d_in[10]};
  const float* asrc[3] = {(const float*)d_in[3],  (const float*)d_in[7],  (const float*)d_in[11]};
  const float* adst[3] = {(const float*)d_in[4],  (const float*)d_in[8],  (const float*)d_in[12]};
  const float* bia[3]  = {(const float*)d_in[5],  (const float*)d_in[9],  (const float*)d_in[13]};
  const float* fw1 = (const float*)d_in[14]; const float* fb1 = (const float*)d_in[15];
  const float* fw2 = (const float*)d_in[16]; const float* fb2 = (const float*)d_in[17];
  const float* fw3 = (const float*)d_in[18]; const float* fb3 = (const float*)d_in[19];
  const float* fw4 = (const float*)d_in[20]; const float* fb4 = (const float*)d_in[21];
  float* out = (float*)d_out;

  // workspace carve-out (all sizes multiples of 256B)
  char* w = (char*)d_ws;
  const size_t BN152 = (size_t)BB * NN * HH;               // 9,961,472 elem
  float* x1   = (float*)w; w += BN152 * 4;
  float* x2   = (float*)w; w += BN152 * 4;
  float* x3   = (float*)w; w += BN152 * 4;
  float* htmp = (float*)w; w += BN152 * 4;
  __bf16* Abuf = (__bf16*)w; w += (size_t)BB * NN * HPAD * 2;
  __bf16* Wt   = (__bf16*)w; w += (size_t)HPAD * HPAD * 2;
  float* hs = (float*)w; w += (size_t)BB * NN * 4;
  float* hd = (float*)w; w += (size_t)BB * NN * 4;
  int* counts  = (int*)w; w += (size_t)BB * NN * 4;
  int* offsets = (int*)w; w += (size_t)BB * OFFS * 4;
  int* cursor  = (int*)w; w += (size_t)BB * NN * 4;
  int* srcl    = (int*)w; w += (size_t)BB * EN * 4;

  const int T = 256;
  // ---- CSR (shared by all 3 layers)
  k_init_counts<<<(BB * NN + T - 1) / T, T, 0, stream>>>(counts);
  k_count_edges<<<(BB * EE + T - 1) / T, T, 0, stream>>>(ei, counts);
  k_scan<<<BB, NN, 0, stream>>>(counts, offsets, cursor);
  k_fill_self<<<(BB * NN + T - 1) / T, T, 0, stream>>>(cursor, srcl);
  k_fill_edges<<<(BB * EE + T - 1) / T, T, 0, stream>>>(ei, cursor, srcl);

  const int gemmBlocks = BB * 8;                           // 8 waves/block, 8 m-tiles each
  const int nodeWaveBlocks = BB * NN / 8;                  // 8192

  float* layerOut[3] = {x1, x2, x3};
  const float* stageAux[3] = {x, x1, x2};
  int   mode[3]  = {0, 1, 2};
  int   Kpad3[3] = {32, HPAD, HPAD};
  int   Kin3[3]  = {3, HH, 155};

  for (int l = 0; l < 3; ++l) {
    int Kpad = Kpad3[l];
    int Ktiles = Kpad / 32;
    k_conv_w<<<(HPAD * Kpad + T - 1) / T, T, 0, stream>>>(W[l], Wt, Kin3[l], Kpad);
    size_t stTot = (size_t)BB * NN * Kpad;
    k_stage<<<(unsigned)((stTot + T - 1) / T), T, 0, stream>>>(x, stageAux[l], Abuf, mode[l], Kpad);
    k_gemm<<<gemmBlocks, T, 0, stream>>>(Abuf, Wt, htmp, Kpad, Ktiles);
    k_scores<<<nodeWaveBlocks, T, 0, stream>>>(htmp, asrc[l], adst[l], hs, hd);
    k_aggregate<<<nodeWaveBlocks, T, 0, stream>>>(htmp, hs, hd, offsets, srcl, bia[l],
                                                  layerOut[l]);
  }

  k_head<<<BB, T, 0, stream>>>(x, x1, x2, x3, fw1, fb1, fw2, fb2, fw3, fb3, fw4, fb4, out);
}
